// VectorQuantizerEMA_30013231464712
// MI455X (gfx1250) — matrix operand order repaired and checked by reading, not compiled, not run on hardware
//
#include <hip/hip_runtime.h>
#include <hip/hip_bf16.h>
#include <float.h>
#include <math.h>

// ---------------------------------------------------------------------------
// VQ-VAE EMA vector quantizer, CDNA5 (gfx1250), wave32 + WMMA bf16 path.
//
// inputs : [65536, 64] f32   embeddings : [64, 512] f32
// GEMM 65536x512x64 via v_wmma_f32_16x16x32_bf16 (f32 accumulate).
// argmin(||e||^2 - 2 x.e)  ==  argmax(x.e - ||e||^2/2): the -||e||^2/2 bias is
// loaded into the WMMA C operand so the hardware accumulate does the bias add.
// ---------------------------------------------------------------------------

typedef __attribute__((ext_vector_type(16))) __bf16 v16bf;
typedef __attribute__((ext_vector_type(8)))  float  v8f;

#define NROWS 65536
#define DIM   64
#define KCB   512

// workspace layout (float indices)
#define WS_SSE  0
#define WS_CS   1                 // 512 floats
#define WS_DW   513               // 32768 floats
#define WS_EN   33281             // 512 floats: -0.5*||e_k||^2
#define WS_B    33796             // 32768 bf16 = 16384 floats (16B aligned)
#define WS_ZERO_COUNT 33281       // sse + cs + dw must be zeroed per call

// output layout (float indices), reference tuple order flattened
#define OFF_Q       0             // 4194304
#define OFF_LOSS    4194304       // 1
#define OFF_PERP    4194305       // 1
#define OFF_IDX     4194306       // 65536 (indices stored as float)
#define OFF_NEWEMB  4259842      // 32768
#define OFF_HIDCS   4292610       // 512
#define OFF_HIDDW   4293122       // 32768

#define TILES_PER_WG 32
#define WAVES_PER_WG 8

__global__ __launch_bounds__(256) void vq_zero(float* ws) {
  int i = blockIdx.x * blockDim.x + threadIdx.x;
  if (i < WS_ZERO_COUNT) ws[i] = 0.0f;
}

// Pre-swizzle embeddings into wave32 WMMA B-operand bf16 layout.
// flat bf16 index i = ((kt*32+ct)*32 + lane)*16 + e
// B element: row K = kt*32 + (e>>3)*16 + (lane>>4)*8 + (e&7), column = ct*16 + (lane&15)
__global__ __launch_bounds__(256) void vq_pack(const float* __restrict__ emb,
                                               float* __restrict__ ws) {
  __bf16* bptr = (__bf16*)(ws + WS_B);
  int i = blockIdx.x * blockDim.x + threadIdx.x;
  if (i < 32768) {
    int e    = i & 15;
    int lane = (i >> 4) & 31;
    int ct   = (i >> 9) & 31;
    int kt   = i >> 14;
    int krow = kt * 32 + (e >> 3) * 16 + (lane >> 4) * 8 + (e & 7);
    int col  = ct * 16 + (lane & 15);
    bptr[i] = (__bf16)emb[krow * KCB + col];
  }
}

// -0.5 * ||e_k||^2 (bias folded into the WMMA C operand)
__global__ __launch_bounds__(256) void vq_enorm(const float* __restrict__ emb,
                                                float* __restrict__ ws) {
  int k = blockIdx.x * blockDim.x + threadIdx.x;
  if (k < KCB) {
    float s = 0.0f;
    for (int d = 0; d < DIM; ++d) { float v = emb[d * KCB + k]; s += v * v; }
    ws[WS_EN + k] = -0.5f * s;
  }
}

__global__ __launch_bounds__(256) void vq_main(const float* __restrict__ flat,
                                               const float* __restrict__ emb,
                                               float* __restrict__ ws,
                                               float* __restrict__ out_q,
                                               float* __restrict__ out_idx) {
  __shared__ v16bf lds_b[2048];   // 64 KB: [(kt*32+ct)*32 + lane]
  __shared__ float lds_en[KCB];   // 2 KB: -0.5*||e_k||^2

  const int tid  = threadIdx.x;
  const int wave = tid >> 5;
  const int lane = tid & 31;

  // cooperative copy of pre-swizzled B + bias into LDS
  {
    const float4* src = (const float4*)(ws + WS_B);
    float4* dst = (float4*)lds_b;
    for (int i = tid; i < 4096; i += 256) dst[i] = src[i];
    for (int i = tid; i < KCB; i += 256) lds_en[i] = ws[WS_EN + i];
  }
  __syncthreads();

  const int h    = lane >> 4;     // half-wave
  const int cgrp = lane & 15;     // this lane's column-within-tile (C layout)

  float* ws_dw = ws + WS_DW;
  float* ws_cs = ws + WS_CS;
  float sse_acc = 0.0f;

  for (int t = 0; t < TILES_PER_WG / WAVES_PER_WG; ++t) {
    const int tile = blockIdx.x * TILES_PER_WG + t * WAVES_PER_WG + wave;
    const int r0 = tile * 16;

    // ---- load A tile (16 rows x 64) as two K=32 bf16 operands -------------
    // A layout: lane holds row M = lane&15; element e -> K = (e>=8?16:0)+h*8+(e&7)
    const float* arow = flat + (size_t)(r0 + cgrp) * DIM;
    v16bf a0, a1;
#pragma unroll
    for (int j = 0; j < 8; ++j) {
      a0[j]     = (__bf16)arow[h * 8 + j];
      a0[8 + j] = (__bf16)arow[16 + h * 8 + j];
      a1[j]     = (__bf16)arow[32 + h * 8 + j];
      a1[8 + j] = (__bf16)arow[48 + h * 8 + j];
    }

    float bestv[8];                 // running MAX of (x.e - ||e||^2/2)
    int   besti[8];
#pragma unroll
    for (int j = 0; j < 8; ++j) { bestv[j] = -FLT_MAX; besti[j] = 0; }

    // ---- 32 column tiles x (2 x v_wmma_f32_16x16x32_bf16) -----------------
#pragma unroll 4
    for (int ct = 0; ct < 32; ++ct) {
      const int col = ct * 16 + cgrp;
      const float bias = lds_en[col];
      v8f acc;
#pragma unroll
      for (int j = 0; j < 8; ++j) acc[j] = bias;   // C := -||e||^2/2
      v16bf b0 = lds_b[ct * 32 + lane];
      v16bf b1 = lds_b[(32 + ct) * 32 + lane];
      acc = __builtin_amdgcn_wmma_f32_16x16x32_bf16(false, a0, false, b0,
                                                    (short)0, acc, false, false);
      acc = __builtin_amdgcn_wmma_f32_16x16x32_bf16(false, a1, false, b1,
                                                    (short)0, acc, false, false);
#pragma unroll
      for (int j = 0; j < 8; ++j) {                // rowslot j -> row h*8+j
        float s = acc[j];
        if (s > bestv[j]) { bestv[j] = s; besti[j] = col; }
      }
    }

    // ---- argmax across the 16 lanes of each half-wave ---------------------
#pragma unroll
    for (int j = 0; j < 8; ++j) {
#pragma unroll
      for (int off = 1; off < 16; off <<= 1) {
        float ov = __shfl_xor(bestv[j], off, 32);
        int   oi = __shfl_xor(besti[j], off, 32);
        if (ov > bestv[j] || (ov == bestv[j] && oi < besti[j])) {
          bestv[j] = ov; besti[j] = oi;
        }
      }
    }

    // ---- per-row outputs: gather code, STE write, sse, dw/cs scatter ------
#pragma unroll
    for (int m = 0; m < 16; ++m) {
      const int slot = m & 7;
      const int src  = (m < 8) ? 0 : 16;
      const int idx  = __shfl(besti[slot], src, 32);
      const int n    = r0 + m;
      const size_t base = (size_t)n * DIM;
      float x0 = flat[base + lane];
      float x1 = flat[base + 32 + lane];
      float q0 = emb[lane * KCB + idx];
      float q1 = emb[(32 + lane) * KCB + idx];
      out_q[base + lane]      = q0;   // quantized_st forward value == quantized
      out_q[base + 32 + lane] = q1;
      float d0 = q0 - x0, d1 = q1 - x1;
      sse_acc += d0 * d0 + d1 * d1;
      atomicAdd(&ws_dw[lane * KCB + idx], x0);
      atomicAdd(&ws_dw[(32 + lane) * KCB + idx], x1);
      if (lane == 0) {
        out_idx[n] = (float)idx;
        atomicAdd(&ws_cs[idx], 1.0f);
      }
    }
  }

  // wave-level SSE reduction, one atomic per wave
#pragma unroll
  for (int off = 16; off >= 1; off >>= 1) sse_acc += __shfl_xor(sse_acc, off, 32);
  if (lane == 0) atomicAdd(ws + WS_SSE, sse_acc);
}

__global__ __launch_bounds__(512) void vq_finalize(const float* __restrict__ ema_cs,
                                                   const float* __restrict__ ema_dw,
                                                   const int* __restrict__ counter,
                                                   const float* __restrict__ ws,
                                                   float* __restrict__ out) {
  __shared__ float s_stable[KCB];
  __shared__ float s_red[16];
  __shared__ float s_red2[16];

  const int tid = threadIdx.x;
  const float debias = 1.0f - powf(0.99f, (float)(counter[0] + 1));

  float cs  = ws[WS_CS + tid];
  float hid = ema_cs[tid] * 0.99f + cs * 0.01f;
  out[OFF_HIDCS + tid] = hid;
  float upd = hid / debias;

  float avg  = cs * (1.0f / (float)NROWS);
  float term = avg * logf(avg + 1e-10f);

  // two wave reductions then cross-wave via LDS
  float v = upd, w = term;
#pragma unroll
  for (int off = 16; off >= 1; off >>= 1) {
    v += __shfl_xor(v, off, 32);
    w += __shfl_xor(w, off, 32);
  }
  if ((tid & 31) == 0) { s_red[tid >> 5] = v; s_red2[tid >> 5] = w; }
  __syncthreads();
  if (tid == 0) {
    float sv = 0.0f, sw = 0.0f;
    for (int i = 0; i < 16; ++i) { sv += s_red[i]; sw += s_red2[i]; }
    s_red[0]  = sv;
    out[OFF_PERP] = expf(-sw);
    out[OFF_LOSS] = 0.25f * ws[WS_SSE] / ((float)NROWS * (float)DIM);
  }
  __syncthreads();
  const float ntot = s_red[0];
  s_stable[tid] = (upd + 1e-5f) / (ntot + (float)KCB * 1e-5f) * ntot;
  __syncthreads();

  for (int i = tid; i < DIM * KCB; i += 512) {
    float hdw = ema_dw[i] * 0.99f + ws[WS_DW + i] * 0.01f;
    out[OFF_HIDDW + i]  = hdw;
    out[OFF_NEWEMB + i] = (hdw / debias) / s_stable[i & (KCB - 1)];
  }
}

extern "C" void kernel_launch(void* const* d_in, const int* in_sizes, int n_in,
                              void* d_out, int out_size, void* d_ws, size_t ws_size,
                              hipStream_t stream) {
  (void)in_sizes; (void)n_in; (void)out_size; (void)ws_size;
  const float* flat   = (const float*)d_in[0];  // [65536, 64]
  const float* emb    = (const float*)d_in[1];  // [64, 512]
  const float* ema_cs = (const float*)d_in[2];  // [512]
  const float* ema_dw = (const float*)d_in[3];  // [64, 512]
  const int*   ctr    = (const int*)d_in[4];    // scalar
  float* out = (float*)d_out;
  float* ws  = (float*)d_ws;                    // needs ~201 KB

  vq_zero  <<<(WS_ZERO_COUNT + 255) / 256, 256, 0, stream>>>(ws);
  vq_pack  <<<128, 256, 0, stream>>>(emb, ws);
  vq_enorm <<<2,   256, 0, stream>>>(emb, ws);
  vq_main  <<<(NROWS / 16) / TILES_PER_WG, 256, 0, stream>>>(
      flat, emb, ws, out + OFF_Q, out + OFF_IDX);
  vq_finalize<<<1, 512, 0, stream>>>(ema_cs, ema_dw, ctr, ws, out);
}
